// MHSelfAttention_52140902973547
// MI455X (gfx1250) — matrix operand order repaired and checked by reading, not compiled, not run on hardware
//
#include <hip/hip_runtime.h>

typedef __attribute__((ext_vector_type(16))) __bf16 bf16x16;
typedef __attribute__((ext_vector_type(8)))  float  f32x8;
typedef unsigned int u32x4 __attribute__((ext_vector_type(4)));
typedef int          i32x4 __attribute__((ext_vector_type(4)));
typedef int          i32x8 __attribute__((ext_vector_type(8)));

union FragB16 { uint4 q[2]; bf16x16 v; };

#define HID  1024
#define NQKV 3072
#define SEQ  2048
#define NBH  64           // 4 batches * 16 heads
#define QKV_ELEMS ((size_t)NBH * SEQ * 64)   // per Q/K/V tensor, bf16 elements

#define HAVE_TDM __has_builtin(__builtin_amdgcn_tensor_load_to_lds)

__device__ __forceinline__ unsigned short f2bf(float f) {
    union { float f; unsigned u; } x; x.f = f;
    unsigned r = x.u + 0x7FFFu + ((x.u >> 16) & 1u);   // round-to-nearest-even
    return (unsigned short)(r >> 16);
}
__device__ __forceinline__ unsigned pack2bf(float lo, float hi) {
    return (unsigned)f2bf(lo) | ((unsigned)f2bf(hi) << 16);
}

// ---------------------------------------------------------------------------
// TDM: DMA a 2D bf16 tile (tile_d0 x tile_d1, innermost contiguous) from
// global memory into LDS. Descriptor per CDNA5 ISA.
// ---------------------------------------------------------------------------
__device__ __forceinline__ void tdm_load_2d_bf16(
        unsigned lds_off, const void* gptr,
        unsigned tile_d0, unsigned tile_d1,
        unsigned tensor_d0, unsigned tensor_d1, unsigned stride0) {
#if HAVE_TDM
    unsigned long long ga = (unsigned long long)gptr;
    u32x4 g0;
    g0[0] = 1u;                                   // count=1, user mode, no gather
    g0[1] = lds_off;                              // lds_addr (bytes)
    g0[2] = (unsigned)(ga & 0xFFFFFFFFu);         // global_addr[31:0]
    g0[3] = (unsigned)((ga >> 32) & 0x01FFFFFFu)  // global_addr[56:32]
          | (2u << 30);                           // type = 2 ("image")
    i32x8 g1;
    g1[0] = (int)(1u << 16);                      // data_size=1 (2 bytes), mask=0
    g1[1] = (int)((tensor_d0 & 0xFFFFu) << 16);   // tensor_dim0[15:0] @ bits 63:48
    g1[2] = (int)(((tensor_d0 >> 16) & 0xFFFFu) | ((tensor_d1 & 0xFFFFu) << 16));
    g1[3] = (int)(((tensor_d1 >> 16) & 0xFFFFu) | ((tile_d0 & 0xFFFFu) << 16));
    g1[4] = (int)(tile_d1 & 0xFFFFu);             // tile_dim1; tile_dim2=0 (unused)
    g1[5] = (int)stride0;                         // tensor_dim0_stride[31:0]
    g1[6] = 0;                                    // stride0[47:32], stride1[15:0]
    g1[7] = 0;
    i32x4 zz = {0, 0, 0, 0};
#if defined(__clang_major__) && (__clang_major__ >= 23)
    i32x8 z8 = {0, 0, 0, 0, 0, 0, 0, 0};
    __builtin_amdgcn_tensor_load_to_lds(g0, g1, zz, zz, z8, 0);
#else
    __builtin_amdgcn_tensor_load_to_lds(g0, g1, zz, zz, 0);
#endif
#endif
}

// ---------------------------------------------------------------------------
// Kernel 1: QKV projection GEMM (fp32 in, bf16 out, WMMA bf16 / f32 acc)
// Block tile 128(M) x 64(N), K-step 32. 8 waves: 4(M) x 2(N), 32x32 each.
// Output scattered into [b][head][s][d] bf16 workspace; Q pre-scaled by 1/32.
// ---------------------------------------------------------------------------
__global__ __launch_bounds__(256) void qkv_gemm_kernel(
        const float* __restrict__ emb, const float* __restrict__ w,
        unsigned short* __restrict__ wsQ, unsigned short* __restrict__ wsK,
        unsigned short* __restrict__ wsV) {
    __shared__ __align__(16) unsigned short ldsA[128 * 32];   // [row][k] bf16
    __shared__ __align__(16) unsigned short ldsB[64 * 32];    // [n][k]  bf16 (transposed)

    const int tid  = threadIdx.x;
    const int lane = tid & 31;
    const int wid  = tid >> 5;
    const int l15  = lane & 15;
    const int half = (lane < 16) ? 0 : 1;
    const int m0 = blockIdx.y * 128;
    const int n0 = blockIdx.x * 64;
    const int wm = wid & 3;    // 0..3 -> M offset wm*32
    const int wn = wid >> 2;   // 0..1 -> N offset wn*32

    f32x8 acc[2][2] = {};

    for (int k0 = 0; k0 < HID; k0 += 32) {
        __syncthreads();
        // ---- stage A: 128 rows x 16 k-pairs (fp32 -> packed bf16) ----
        {
            const float2* E2 = (const float2*)emb;
            unsigned* A32 = (unsigned*)ldsA;
#pragma unroll
            for (int i = 0; i < 8; ++i) {
                int task = tid + i * 256;          // 0..2047
                int r  = task >> 4;
                int kp = task & 15;
                float2 v = E2[(size_t)(m0 + r) * (HID / 2) + (k0 >> 1) + kp];
                A32[r * 16 + kp] = pack2bf(v.x, v.y);
            }
        }
        // ---- stage B transposed: 64 cols x 16 k-pairs ----
        {
            unsigned* B32 = (unsigned*)ldsB;
#pragma unroll
            for (int i = 0; i < 4; ++i) {
                int task = tid + i * 256;          // 0..1023
                int c  = task >> 4;
                int kp = task & 15;
                float lo = w[(size_t)(k0 + 2 * kp) * NQKV + n0 + c];
                float hi = w[(size_t)(k0 + 2 * kp + 1) * NQKV + n0 + c];
                B32[c * 16 + kp] = pack2bf(lo, hi);
            }
        }
        // ---- prefetch next K-step tiles (global_prefetch_b8) ----
        if (k0 + 32 < HID) {
            int r = tid >> 1;                                  // 0..127: A rows
            __builtin_prefetch(emb + (size_t)(m0 + r) * HID + k0 + 32 + (tid & 1) * 16, 0, 0);
            int kr = tid & 31;                                 // 0..31: B rows
            __builtin_prefetch(w + (size_t)(k0 + 32 + kr) * NQKV + n0 + (tid >> 5) * 8, 0, 0);
        }
        __syncthreads();

        // ---- fragments: each is two b128 LDS reads per lane ----
        FragB16 af[2], bfg[2];
        const uint4* A4 = (const uint4*)ldsA;
        const uint4* B4 = (const uint4*)ldsB;
#pragma unroll
        for (int t = 0; t < 2; ++t) {
            int r = wm * 32 + t * 16 + l15;        // A row, stride 64B
            af[t].q[0] = A4[r * 4 + half];
            af[t].q[1] = A4[r * 4 + 2 + half];
            int c = wn * 32 + t * 16 + l15;        // B col (N), stride 64B
            bfg[t].q[0] = B4[c * 4 + half];
            bfg[t].q[1] = B4[c * 4 + 2 + half];
        }
#pragma unroll
        for (int ta = 0; ta < 2; ++ta)
#pragma unroll
            for (int tb = 0; tb < 2; ++tb)
                acc[ta][tb] = __builtin_amdgcn_wmma_f32_16x16x32_bf16(
                    false, af[ta].v, false, bfg[tb].v, (short)0, acc[ta][tb],
                    false, false);
    }

    // ---- scatter-store into [b][head][s][d] bf16 workspace ----
    const int mrow = half * 8;
#pragma unroll
    for (int ta = 0; ta < 2; ++ta) {
#pragma unroll
        for (int tb = 0; tb < 2; ++tb) {
            int n    = n0 + wn * 32 + tb * 16 + l15;
            int qkv  = n >> 10;
            int head = (n >> 6) & 15;
            int d    = n & 63;
            unsigned short* tgt = (qkv == 0) ? wsQ : ((qkv == 1) ? wsK : wsV);
            float scl = (qkv == 0) ? 0.03125f : 1.0f;   // 1/sqrt(1024) baked into Q
#pragma unroll
            for (int v = 0; v < 8; ++v) {
                int m    = m0 + wm * 32 + ta * 16 + mrow + v;
                int b    = m >> 11;        // /2048
                int sidx = m & 2047;
                size_t off = (((size_t)(b * 16 + head)) * SEQ + sidx) * 64 + d;
                tgt[off] = f2bf(acc[ta][tb][v] * scl);
            }
        }
    }
}

// ---------------------------------------------------------------------------
// Kernel 2: flash attention per (b,head). Block = 128 q rows (8 waves x 16).
// K tiles staged via TDM (double-buffered, TENSORcnt-synced); V staged with a
// manual transpose; 4 QK^T + 4 PV WMMAs per 32-key block per wave.
// ---------------------------------------------------------------------------
__global__ __launch_bounds__(256) void attn_kernel(
        const unsigned short* __restrict__ wsQ,
        const unsigned short* __restrict__ wsK,
        const unsigned short* __restrict__ wsV,
        float* __restrict__ out) {
    __shared__ __align__(16) unsigned short ldsK[2][32 * 64];   // raw [key][d], dbl-buf
    __shared__ __align__(16) unsigned short ldsVt[64 * 32];     // [d][key] bf16
    __shared__ __align__(16) unsigned short ldsP[8][16 * 32];   // per-wave P tile

    const int tid  = threadIdx.x;
    const int lane = tid & 31;
    const int wid  = tid >> 5;
    const int l15  = lane & 15;
    const int half = (lane < 16) ? 0 : 1;
    const int bh    = blockIdx.y;                  // b*16 + head
    const int qbase = blockIdx.x * 128 + wid * 16;
    const size_t base = (size_t)bh * SEQ * 64;     // element offset into Q/K/V

    const unsigned short* Kbase = wsK + base;

    // ---- load Q fragments once (A-layout, contiguous b128 reads) ----
    FragB16 qf[2];
    {
        const uint4* Q4 = (const uint4*)(wsQ + base);
        int r = qbase + l15;                       // row stride = 128B = 8 uint4
        qf[0].q[0] = Q4[r * 8 + half];
        qf[0].q[1] = Q4[r * 8 + 2 + half];
        qf[1].q[0] = Q4[r * 8 + 4 + half];
        qf[1].q[1] = Q4[r * 8 + 6 + half];
    }

    f32x8 o[4] = {};
    float mrun[8], lrun[8];
#pragma unroll
    for (int v = 0; v < 8; ++v) { mrun[v] = -1e30f; lrun[v] = 0.0f; }

    const unsigned* V32 = (const unsigned*)(wsV + base);

    int buf = 0;
#if HAVE_TDM
    const unsigned kOff0 = (unsigned)(unsigned long long)(const void*)&ldsK[0][0];
    const unsigned kOff1 = (unsigned)(unsigned long long)(const void*)&ldsK[1][0];
    if (wid == 0)   // prologue: DMA first K tile
        tdm_load_2d_bf16(kOff0, Kbase, 64, 32, 64, SEQ, 64);
#endif

    for (int kb = 0; kb < SEQ; kb += 32) {
#if HAVE_TDM
        if (wid == 0) __builtin_amdgcn_s_wait_tensorcnt(0);   // K[buf] landed
        __syncthreads();                                      // publish; prior reads done
        if (wid == 0 && (kb + 32 < SEQ))                      // DMA next tile into buf^1
            tdm_load_2d_bf16(buf ? kOff0 : kOff1,
                             Kbase + (size_t)(kb + 32) * 64, 64, 32, 64, SEQ, 64);
#else
        __syncthreads();
        ((uint4*)&ldsK[0][0])[tid] = ((const uint4*)Kbase)[kb * 8 + tid];  // 4KB copy
#endif
        // ---- cooperative stage of V block transposed: [d][key] ----
#pragma unroll
        for (int i = 0; i < 4; ++i) {
            int task = tid + i * 256;              // 0..1023: key(32) x dpair(32)
            int key = task >> 5;
            int dp  = task & 31;
            unsigned val = V32[(size_t)(kb + key) * 32 + dp];
            ldsVt[(2 * dp)     * 32 + key] = (unsigned short)(val & 0xFFFFu);
            ldsVt[(2 * dp + 1) * 32 + key] = (unsigned short)(val >> 16);
        }
        // prefetch next V tile
        if (kb + 32 < SEQ)
            __builtin_prefetch(wsV + base + (size_t)(kb + 32) * 64 + tid * 8, 0, 0);
        __syncthreads();

        // ---- S = Q K^T for two 16-key tiles (B frags from LDS K tile) ----
        const uint4* KT4 = (const uint4*)&ldsK[buf][0];
        f32x8 s0 = {}, s1 = {};
        FragB16 kf;
        kf.q[0] = KT4[l15 * 8 + half];
        kf.q[1] = KT4[l15 * 8 + 2 + half];
        s0 = __builtin_amdgcn_wmma_f32_16x16x32_bf16(false, qf[0].v, false, kf.v, (short)0, s0, false, false);
        kf.q[0] = KT4[l15 * 8 + 4 + half];
        kf.q[1] = KT4[l15 * 8 + 6 + half];
        s0 = __builtin_amdgcn_wmma_f32_16x16x32_bf16(false, qf[1].v, false, kf.v, (short)0, s0, false, false);
        kf.q[0] = KT4[(16 + l15) * 8 + half];
        kf.q[1] = KT4[(16 + l15) * 8 + 2 + half];
        s1 = __builtin_amdgcn_wmma_f32_16x16x32_bf16(false, qf[0].v, false, kf.v, (short)0, s1, false, false);
        kf.q[0] = KT4[(16 + l15) * 8 + 4 + half];
        kf.q[1] = KT4[(16 + l15) * 8 + 6 + half];
        s1 = __builtin_amdgcn_wmma_f32_16x16x32_bf16(false, qf[1].v, false, kf.v, (short)0, s1, false, false);

        // ---- online softmax (row reductions across 16-lane halves) ----
#pragma unroll
        for (int v = 0; v < 8; ++v) {
            float tmax = fmaxf(s0[v], s1[v]);
#pragma unroll
            for (int off = 8; off > 0; off >>= 1)
                tmax = fmaxf(tmax, __shfl_xor(tmax, off, 16));
            float mnew = fmaxf(mrun[v], tmax);
            float c  = __expf(mrun[v] - mnew);
            float p0 = __expf(s0[v] - mnew);
            float p1 = __expf(s1[v] - mnew);
            float rs = p0 + p1;
#pragma unroll
            for (int off = 8; off > 0; off >>= 1)
                rs += __shfl_xor(rs, off, 16);
            lrun[v] = lrun[v] * c + rs;
            mrun[v] = mnew;
#pragma unroll
            for (int t = 0; t < 4; ++t) o[t][v] *= c;
            // write P (C-layout -> LDS row-major [m][key])
            int m = v + half * 8;
            ldsP[wid][m * 32 + l15]      = f2bf(p0);
            ldsP[wid][m * 32 + 16 + l15] = f2bf(p1);
        }

        // ---- P A-fragment from LDS, V B-fragments from transposed LDS ----
        FragB16 pf;
        const uint4* P4 = (const uint4*)ldsP[wid];
        pf.q[0] = P4[l15 * 4 + half];
        pf.q[1] = P4[l15 * 4 + 2 + half];
        const uint4* Vt4 = (const uint4*)ldsVt;
#pragma unroll
        for (int t = 0; t < 4; ++t) {
            FragB16 vf;
            int d = t * 16 + l15;
            vf.q[0] = Vt4[d * 4 + half];
            vf.q[1] = Vt4[d * 4 + 2 + half];
            o[t] = __builtin_amdgcn_wmma_f32_16x16x32_bf16(
                false, pf.v, false, vf.v, (short)0, o[t], false, false);
        }
#if HAVE_TDM
        buf ^= 1;
#endif
    }

    // ---- finalize: O /= l, scatter fp32 to [b][s][hidden] ----
    const int b    = bh >> 4;
    const int head = bh & 15;
#pragma unroll
    for (int v = 0; v < 8; ++v) {
        float inv = 1.0f / lrun[v];
        int q = qbase + v + half * 8;
#pragma unroll
        for (int t = 0; t < 4; ++t) {
            int d = t * 16 + l15;
            out[((size_t)(b * SEQ + q)) * HID + head * 64 + d] = o[t][v] * inv;
        }
    }
}

extern "C" void kernel_launch(void* const* d_in, const int* in_sizes, int n_in,
                              void* d_out, int out_size, void* d_ws, size_t ws_size,
                              hipStream_t stream) {
    const float* emb = (const float*)d_in[0];   // [4,2048,1024] fp32
    const float* w   = (const float*)d_in[1];   // [1024,3072]   fp32
    float* out = (float*)d_out;                 // [4,2048,1024] fp32

    unsigned short* wsQ = (unsigned short*)d_ws;        // 16 MB each, bf16
    unsigned short* wsK = wsQ + QKV_ELEMS;
    unsigned short* wsV = wsK + QKV_ELEMS;

    dim3 g1(NQKV / 64, (4 * SEQ) / 128);   // (48, 64)
    qkv_gemm_kernel<<<g1, 256, 0, stream>>>(emb, w, wsQ, wsK, wsV);

    dim3 g2(SEQ / 128, NBH);               // (16, 64)
    attn_kernel<<<g2, 256, 0, stream>>>(wsQ, wsK, wsV, out);
}
